// MultiheadDilatedSelfAttention_72971494359476
// MI455X (gfx1250) — matrix operand order repaired and checked by reading, not compiled, not run on hardware
//
#include <hip/hip_runtime.h>
#include <hip/hip_bf16.h>
#include <stdint.h>

// ---------------- types / helpers -------------------------------------------
typedef __attribute__((ext_vector_type(16))) __bf16 v16bf;
typedef __attribute__((ext_vector_type(8)))  float  v8f;

#define DEVINL static __device__ __forceinline__

struct B32x2 { uint4 lo, hi; };

DEVINL v16bf make_frag(uint4 lo, uint4 hi) {
    B32x2 t; t.lo = lo; t.hi = hi;
    return __builtin_bit_cast(v16bf, t);
}

DEVINL unsigned short f2bf(float f) {   // round-to-nearest-even fp32 -> bf16
    unsigned u = __builtin_bit_cast(unsigned, f);
    u += 0x7fffu + ((u >> 16) & 1u);
    return (unsigned short)(u >> 16);
}

DEVINL v8f zero8() { v8f z = {0.f,0.f,0.f,0.f,0.f,0.f,0.f,0.f}; return z; }

union V8u16 { uint4 v; unsigned short u[8]; };

// ---------------- problem constants -----------------------------------------
#define BB    2
#define NN    4096
#define CC    1024
#define HH    16
#define DH    64
#define MTOT  (BB*NN)        /* 8192 */
#define QKVN  (HH*3*DH)      /* 3072 */

// =============================================================================
// Kernel 1: QKV projection.  x(8192x1024) @ Wqkv(1024 x 3072, head-blocked)
// Block tile 128x128, 8 waves, each wave 64x32 (4x2 wmma fragments).
// =============================================================================
__global__ __launch_bounds__(256)
void qkv_gemm_kernel(const float* __restrict__ x,
                     const float* __restrict__ Wqkv,
                     const float* __restrict__ bqkv,
                     unsigned short* __restrict__ qg,
                     unsigned short* __restrict__ kg,
                     unsigned short* __restrict__ vg)
{
    __shared__ unsigned short Ash[128][48];   // [m][k]  (stride 96B, 16B aligned)
    __shared__ unsigned short Bsh[128][48];   // [n][k]  transposed B tile

    const int t    = threadIdx.x;
    const int lane = t & 31;
    const int wave = t >> 5;
    const int bm   = blockIdx.x * 128;
    const int bn   = blockIdx.y * 128;
    const int wm   = (wave & 1) * 64;
    const int wn   = (wave >> 1) * 32;

    v8f acc[4][2];
#pragma unroll
    for (int i = 0; i < 4; ++i)
#pragma unroll
        for (int j = 0; j < 2; ++j) acc[i][j] = zero8();

    for (int kk = 0; kk < CC; kk += 32) {
        __syncthreads();
        // ---- stage A tile (128x32 fp32 -> bf16) ----
#pragma unroll
        for (int i = 0; i < 4; ++i) {
            int fi  = t + 256 * i;            // 0..1023 float4 slots
            int row = fi >> 3;
            int c4  = (fi & 7) * 4;
            const float4 a4 = *(const float4*)(x + (size_t)(bm + row) * CC + kk + c4);
            if (kk + 32 < CC)                 // global_prefetch_b8
                __builtin_prefetch(x + (size_t)(bm + row) * CC + kk + 32 + c4, 0, 0);
            uint2 pk;
            pk.x = (unsigned)f2bf(a4.x) | ((unsigned)f2bf(a4.y) << 16);
            pk.y = (unsigned)f2bf(a4.z) | ((unsigned)f2bf(a4.w) << 16);
            *(uint2*)&Ash[row][c4] = pk;
        }
        // ---- stage B tile (32x128, head-blocked source) transposed ----
#pragma unroll
        for (int i = 0; i < 4; ++i) {
            int fi   = t + 256 * i;
            int krow = fi >> 5;               // 0..31
            int c4   = (fi & 31) * 4;         // 0..124
            int cg   = bn + c4;
            int h    = cg / 192;
            int j    = cg - h * 192;
            const float4 b4 = *(const float4*)(Wqkv + (size_t)h * (CC * 192)
                                               + (size_t)(kk + krow) * 192 + j);
            Bsh[c4 + 0][krow] = f2bf(b4.x);
            Bsh[c4 + 1][krow] = f2bf(b4.y);
            Bsh[c4 + 2][krow] = f2bf(b4.z);
            Bsh[c4 + 3][krow] = f2bf(b4.w);
        }
        __syncthreads();

        // ---- fragments + WMMA ----
        v16bf afr[4];
#pragma unroll
        for (int mf = 0; mf < 4; ++mf) {
            int row = wm + mf * 16 + (lane & 15);
            int off = (lane >> 4) * 8;        // interleaved ISA A layout
            uint4 lo = *(const uint4*)&Ash[row][off];
            uint4 hi = *(const uint4*)&Ash[row][16 + off];
            afr[mf] = make_frag(lo, hi);
        }
        v16bf bfr[2];
#pragma unroll
        for (int nf = 0; nf < 2; ++nf) {
            int col = wn + nf * 16 + (lane & 15);
            int off = (lane >> 4) * 8;        // B: interleaved ISA layout (matches A)
            uint4 lo = *(const uint4*)&Bsh[col][off];
            uint4 hi = *(const uint4*)&Bsh[col][off + 16];
            bfr[nf] = make_frag(lo, hi);
        }
#pragma unroll
        for (int mf = 0; mf < 4; ++mf)
#pragma unroll
            for (int nf = 0; nf < 2; ++nf)
                acc[mf][nf] = __builtin_amdgcn_wmma_f32_16x16x32_bf16(
                    false, afr[mf], false, bfr[nf], (short)0, acc[mf][nf], false, false);
    }

    // ---- epilogue: +bias, bf16, scatter into q/k/v [b][h][n][d] ----
#pragma unroll
    for (int mf = 0; mf < 4; ++mf)
#pragma unroll
        for (int nf = 0; nf < 2; ++nf)
#pragma unroll
            for (int e = 0; e < 8; ++e) {
                int rg = bm + wm + mf * 16 + e + ((lane >> 4) << 3);
                int cg = bn + wn + nf * 16 + (lane & 15);
                int h  = cg / 192, j = cg - h * 192;
                float val = acc[mf][nf][e] + bqkv[h * 192 + j];
                int which = j >> 6, d = j & 63;
                int b = rg >> 12, n = rg & 4095;
                size_t idx = ((size_t)(b * HH + h) * NN + n) * DH + d;
                unsigned short bv = f2bf(val);
                if      (which == 0) qg[idx] = bv;
                else if (which == 1) kg[idx] = bv;
                else                 vg[idx] = bv;
            }
}

// =============================================================================
// Kernel 2: zero-fill num/den accumulators
// =============================================================================
__global__ void zero_kernel(float4* __restrict__ p, long n4)
{
    long i      = (long)blockIdx.x * blockDim.x + threadIdx.x;
    long stride = (long)gridDim.x * blockDim.x;
    float4 z = make_float4(0.f, 0.f, 0.f, 0.f);
    for (; i < n4; i += stride) p[i] = z;
}

// =============================================================================
// Kernel 3: dilated flash attention, one branch (w,r) per launch, m = 1024.
// Block = 1 (b,h,segment,qtile-of-128); 8 waves, each wave owns 16 q rows.
// K tile is staged with the CDNA5 async global->LDS DMA path (ASYNCcnt).
// =============================================================================
__global__ __launch_bounds__(256)
void attn_kernel(const unsigned short* __restrict__ qg,
                 const unsigned short* __restrict__ kg,
                 const unsigned short* __restrict__ vg,
                 float* __restrict__ num,
                 float* __restrict__ den,
                 int r, int S)
{
    __shared__ unsigned short Ksh[128][72];       // [key][d]
    __shared__ unsigned short Vsh[64][136];       // [d][key]   (V transposed)
    __shared__ unsigned short Psh[8][16][136];    // per-wave P staging

    const int t = threadIdx.x, lane = t & 31, wave = t >> 5;
    int blk = blockIdx.x;
    const int QT = 8;                              // 1024 / 128
    int qt = blk % QT;  blk /= QT;
    int s  = blk % S;   blk /= S;
    int bh = blk;                                  // b*H + h, 0..31
    int h  = bh & (HH - 1);
    int off = h & (r - 1);                         // h % r (r is power of 2)
    const int m = 1024;
    const size_t rowbase = (size_t)bh * NN;

    // ---- Q fragments (direct global load in A layout) ----
    int tq = qt * 128 + wave * 16 + (lane & 15);
    int nq = off + r * (s * m + tq);
    v16bf qf[2];
#pragma unroll
    for (int kf = 0; kf < 2; ++kf) {
        const unsigned short* p = qg + (rowbase + nq) * DH + kf * 32 + ((lane >> 4) * 8);
        uint4 lo = *(const uint4*)p;
        uint4 hi = *(const uint4*)(p + 16);
        qf[kf] = make_frag(lo, hi);
    }

    float rmax[8], rsum[8];
#pragma unroll
    for (int e = 0; e < 8; ++e) { rmax[e] = -3.0e38f; rsum[e] = 0.f; }
    v8f o[4];
#pragma unroll
    for (int df = 0; df < 4; ++df) o[df] = zero8();

    for (int kt = 0; kt <= qt; ++kt) {
        __syncthreads();   // all reads of the previous tile retired before overwrite
        // ---- K tile: async DMA global -> LDS (no VGPR round-trip) ----
        // ---- V tile: manual staged transpose (needs element shuffle)  ----
#pragma unroll
        for (int i = 0; i < 4; ++i) {
            int fi = t + 256 * i;                  // 1024 b128 slots
            int kn = fi >> 3;                      // key row 0..127
            int d8 = (fi & 7) * 8;                 // d offset
            int nk = off + r * (s * m + kt * 128 + kn);
            const unsigned short* kp = kg + (rowbase + nk) * DH + d8;
            unsigned ldsK = (unsigned)(size_t)&Ksh[kn][d8];   // LDS byte address
            asm volatile("global_load_async_to_lds_b128 %0, %1, off"
                         :: "v"(ldsK), "v"(kp) : "memory");
            V8u16 vv; vv.v = *(const uint4*)(vg + (rowbase + nk) * DH + d8);
#pragma unroll
            for (int j = 0; j < 8; ++j) Vsh[d8 + j][kn] = vv.u[j];
        }
        asm volatile("s_wait_asynccnt 0x0" ::: "memory");  // K DMA landed in LDS
        __syncthreads();

        // ---- scores S = Q K^T * 1/sqrt(d), causal mask ----
        float sv[8][8];
        float tmax[8];
#pragma unroll
        for (int e = 0; e < 8; ++e) tmax[e] = -3.0e38f;
#pragma unroll
        for (int nf = 0; nf < 8; ++nf) {
            v8f sa = zero8();
            int key = nf * 16 + (lane & 15);
#pragma unroll
            for (int kf = 0; kf < 2; ++kf) {
                int offk = kf * 32 + ((lane >> 4) * 8);
                uint4 lo = *(const uint4*)&Ksh[key][offk];
                uint4 hi = *(const uint4*)&Ksh[key][offk + 16];
                v16bf kb = make_frag(lo, hi);
                sa = __builtin_amdgcn_wmma_f32_16x16x32_bf16(
                    false, qf[kf], false, kb, (short)0, sa, false, false);
            }
            int tk = kt * 128 + key;
#pragma unroll
            for (int e = 0; e < 8; ++e) {
                int tr = qt * 128 + wave * 16 + e + ((lane >> 4) << 3);
                float v = sa[e] * 0.125f;          // 1/sqrt(64)
                if (tk > tr) v = -3.0e38f;
                sv[nf][e] = v;
                tmax[e] = fmaxf(tmax[e], v);
            }
        }
        // row max across the 16 lanes that share a row
#pragma unroll
        for (int e = 0; e < 8; ++e) {
            float v = tmax[e];
            v = fmaxf(v, __shfl_xor(v, 1, 32));
            v = fmaxf(v, __shfl_xor(v, 2, 32));
            v = fmaxf(v, __shfl_xor(v, 4, 32));
            v = fmaxf(v, __shfl_xor(v, 8, 32));
            tmax[e] = v;
        }
        float fac[8];
#pragma unroll
        for (int e = 0; e < 8; ++e) {
            float nm = fmaxf(rmax[e], tmax[e]);
            fac[e] = __expf(rmax[e] - nm);
            rmax[e] = nm;
            rsum[e] *= fac[e];
        }
#pragma unroll
        for (int df = 0; df < 4; ++df)
#pragma unroll
            for (int e = 0; e < 8; ++e) o[df][e] = o[df][e] * fac[e];

        // ---- P = exp(S - max); stage to LDS (C-layout -> A-layout) ----
        float psum[8];
#pragma unroll
        for (int e = 0; e < 8; ++e) psum[e] = 0.f;
#pragma unroll
        for (int nf = 0; nf < 8; ++nf) {
            int colu = nf * 16 + (lane & 15);
#pragma unroll
            for (int e = 0; e < 8; ++e) {
                float p = __expf(sv[nf][e] - rmax[e]);
                psum[e] += p;
                Psh[wave][e + ((lane >> 4) << 3)][colu] = f2bf(p);
            }
        }
#pragma unroll
        for (int e = 0; e < 8; ++e) {
            float v = psum[e];
            v += __shfl_xor(v, 1, 32);
            v += __shfl_xor(v, 2, 32);
            v += __shfl_xor(v, 4, 32);
            v += __shfl_xor(v, 8, 32);
            rsum[e] += v;
        }

        // ---- O += P @ V ----
#pragma unroll
        for (int kf = 0; kf < 4; ++kf) {
            int prow = lane & 15;
            int offp = kf * 32 + ((lane >> 4) * 8);
            uint4 plo = *(const uint4*)&Psh[wave][prow][offp];
            uint4 phi = *(const uint4*)&Psh[wave][prow][offp + 16];
            v16bf pf = make_frag(plo, phi);
#pragma unroll
            for (int df = 0; df < 4; ++df) {
                int dc   = df * 16 + (lane & 15);
                int offv = kf * 32 + ((lane >> 4) * 8);
                uint4 vlo = *(const uint4*)&Vsh[dc][offv];
                uint4 vhi = *(const uint4*)&Vsh[dc][offv + 16];
                v16bf vb = make_frag(vlo, vhi);
                o[df] = __builtin_amdgcn_wmma_f32_16x16x32_bf16(
                    false, pf, false, vb, (short)0, o[df], false, false);
            }
        }
    }

    // ---- epilogue: LongNet weighted accumulation  num += o*e^max, den += sum*e^max
    float ws[8];
#pragma unroll
    for (int e = 0; e < 8; ++e) ws[e] = __expf(rmax[e]);
#pragma unroll
    for (int df = 0; df < 4; ++df)
#pragma unroll
        for (int e = 0; e < 8; ++e) {
            int tr = qt * 128 + wave * 16 + e + ((lane >> 4) << 3);
            int nr = off + r * (s * m + tr);
            int dc = df * 16 + (lane & 15);
            float* p = num + (rowbase + nr) * DH + dc;
            *p += o[df][e] * ws[e];
        }
    if ((lane & 15) == 0) {
#pragma unroll
        for (int e = 0; e < 8; ++e) {
            int tr = qt * 128 + wave * 16 + e + ((lane >> 4) << 3);
            int nr = off + r * (s * m + tr);
            den[rowbase + nr] += rsum[e] * ws[e];
        }
    }
}

// =============================================================================
// Kernel 4: combine heads (num/den) fused into output GEMM @ Wo + bo
// =============================================================================
__global__ __launch_bounds__(256)
void out_gemm_kernel(const float* __restrict__ num,
                     const float* __restrict__ den,
                     const float* __restrict__ Wo,
                     const float* __restrict__ bo,
                     float* __restrict__ y)
{
    __shared__ unsigned short Ash[128][48];
    __shared__ unsigned short Bsh[128][48];

    const int t    = threadIdx.x;
    const int lane = t & 31;
    const int wave = t >> 5;
    const int bm   = blockIdx.x * 128;
    const int bn   = blockIdx.y * 128;
    const int wm   = (wave & 1) * 64;
    const int wn   = (wave >> 1) * 32;

    v8f acc[4][2];
#pragma unroll
    for (int i = 0; i < 4; ++i)
#pragma unroll
        for (int j = 0; j < 2; ++j) acc[i][j] = zero8();

    for (int kk = 0; kk < CC; kk += 32) {
        __syncthreads();
        // ---- A tile: fused heads-concat + num/den division ----
#pragma unroll
        for (int i = 0; i < 4; ++i) {
            int fi  = t + 256 * i;
            int row = fi >> 3;
            int c4  = (fi & 7) * 4;
            int rg  = bm + row;
            int b   = rg >> 12, n = rg & 4095;
            int c   = kk + c4;
            int h   = c >> 6, d = c & 63;
            size_t base = (size_t)(b * HH + h) * NN + n;
            float4 a4 = *(const float4*)(num + base * DH + d);
            float inv = 1.0f / den[base];          // r=1 branch guarantees den>0
            uint2 pk;
            pk.x = (unsigned)f2bf(a4.x * inv) | ((unsigned)f2bf(a4.y * inv) << 16);
            pk.y = (unsigned)f2bf(a4.z * inv) | ((unsigned)f2bf(a4.w * inv) << 16);
            *(uint2*)&Ash[row][c4] = pk;
        }
        // ---- B tile: Wo (1024x1024 row-major), transposed into LDS ----
#pragma unroll
        for (int i = 0; i < 4; ++i) {
            int fi   = t + 256 * i;
            int krow = fi >> 5;
            int c4   = (fi & 31) * 4;
            const float4 b4 = *(const float4*)(Wo + (size_t)(kk + krow) * CC + bn + c4);
            Bsh[c4 + 0][krow] = f2bf(b4.x);
            Bsh[c4 + 1][krow] = f2bf(b4.y);
            Bsh[c4 + 2][krow] = f2bf(b4.z);
            Bsh[c4 + 3][krow] = f2bf(b4.w);
        }
        __syncthreads();

        v16bf afr[4];
#pragma unroll
        for (int mf = 0; mf < 4; ++mf) {
            int row = wm + mf * 16 + (lane & 15);
            int off = (lane >> 4) * 8;
            uint4 lo = *(const uint4*)&Ash[row][off];
            uint4 hi = *(const uint4*)&Ash[row][16 + off];
            afr[mf] = make_frag(lo, hi);
        }
        v16bf bfr[2];
#pragma unroll
        for (int nf = 0; nf < 2; ++nf) {
            int col = wn + nf * 16 + (lane & 15);
            int off = (lane >> 4) * 8;
            uint4 lo = *(const uint4*)&Bsh[col][off];
            uint4 hi = *(const uint4*)&Bsh[col][off + 16];
            bfr[nf] = make_frag(lo, hi);
        }
#pragma unroll
        for (int mf = 0; mf < 4; ++mf)
#pragma unroll
            for (int nf = 0; nf < 2; ++nf)
                acc[mf][nf] = __builtin_amdgcn_wmma_f32_16x16x32_bf16(
                    false, afr[mf], false, bfr[nf], (short)0, acc[mf][nf], false, false);
    }

#pragma unroll
    for (int mf = 0; mf < 4; ++mf)
#pragma unroll
        for (int nf = 0; nf < 2; ++nf)
#pragma unroll
            for (int e = 0; e < 8; ++e) {
                int rg = bm + wm + mf * 16 + e + ((lane >> 4) << 3);
                int cg = bn + wn + nf * 16 + (lane & 15);
                y[(size_t)rg * CC + cg] = acc[mf][nf][e] + bo[cg];
            }
}

// =============================================================================
// Launcher
// =============================================================================
extern "C" void kernel_launch(void* const* d_in, const int* in_sizes, int n_in,
                              void* d_out, int out_size, void* d_ws, size_t ws_size,
                              hipStream_t stream)
{
    (void)in_sizes; (void)n_in; (void)out_size; (void)ws_size;

    const float* x    = (const float*)d_in[0];   // (B,N,C)
    const float* Wqkv = (const float*)d_in[1];   // (H,C,3*Dh)
    const float* bqkv = (const float*)d_in[2];   // (H,3*Dh)
    const float* Wo   = (const float*)d_in[3];   // (C,C)
    const float* bo   = (const float*)d_in[4];   // (C,)
    float* y = (float*)d_out;

    // workspace layout: q/k/v bf16 (16MB each) | num f32 (32MB) | den f32 (0.5MB)
    const size_t qkvElems = (size_t)BB * HH * NN * DH;   // 8,388,608
    unsigned short* qg = (unsigned short*)d_ws;
    unsigned short* kg = qg + qkvElems;
    unsigned short* vg = kg + qkvElems;
    float* num = (float*)(vg + qkvElems);
    float* den = num + qkvElems;

    // 1) QKV projection GEMM
    qkv_gemm_kernel<<<dim3(MTOT / 128, QKVN / 128), 256, 0, stream>>>(
        x, Wqkv, bqkv, qg, kg, vg);

    // 2) zero num/den (contiguous region)
    long zf = (long)(qkvElems + (size_t)BB * HH * NN);   // floats, /4 exact
    zero_kernel<<<2048, 256, 0, stream>>>((float4*)num, zf / 4);

    // 3) dilated attention branches (w,r) = (1024,1),(2048,2),(4096,4); m=1024
    attn_kernel<<<BB * HH * 4 * 8, 256, 0, stream>>>(qg, kg, vg, num, den, 1, 4);
    attn_kernel<<<BB * HH * 2 * 8, 256, 0, stream>>>(qg, kg, vg, num, den, 2, 2);
    attn_kernel<<<BB * HH * 1 * 8, 256, 0, stream>>>(qg, kg, vg, num, den, 4, 1);

    // 4) combine heads + output projection
    out_gemm_kernel<<<dim3(MTOT / 128, CC / 128), 256, 0, stream>>>(
        num, den, Wo, bo, y);
}
